// NGCF_52458730553699
// MI455X (gfx1250) — matrix operand order repaired and checked by reading, not compiled, not run on hardware
//
#include <hip/hip_runtime.h>

typedef __attribute__((ext_vector_type(16))) _Float16 v16h;
typedef __attribute__((ext_vector_type(8)))  float    v8f;
typedef __attribute__((ext_vector_type(4)))  float    v4f;

#define NUM_USERS 50000
#define NUM_ITEMS 50000
#define N_NODES   (NUM_USERS + NUM_ITEMS)
#define DIM       64
#define OUT_DIM   256
#define N_LAYERS  3
#define EPS_NRM   1e-12f
#define WPACK_ELEMS (N_LAYERS * 2 * 2 * 4 * 32 * 16)   // layer,mat,kc,nt,lane,halves

// ---------------------------------------------------------------------------
// Kernel 1: ego = concat(user_emb, item_emb); also write layer-0 (raw) slice
// of the output (columns 0..63).
// ---------------------------------------------------------------------------
__global__ void init_ego_kernel(const float* __restrict__ ue,
                                const float* __restrict__ ie,
                                float* __restrict__ ego,
                                float* __restrict__ out) {
    int gid = blockIdx.x * blockDim.x + threadIdx.x;
    if (gid >= N_NODES * DIM) return;
    int n = gid >> 6;
    int d = gid & 63;
    float v = (n < NUM_USERS) ? ue[gid] : ie[gid - NUM_USERS * DIM];
    ego[gid] = v;
    out[(size_t)n * OUT_DIM + d] = v;
}

// ---------------------------------------------------------------------------
// Kernel 2: pre-pack W_gcn / W_bi (f32 [3][64][64]) into f16 in the per-lane
// B-matrix layout for V_WMMA_F32_16X16X32_F16. Per ISA §7.12.4 (f16 B-matrix
// table; first 8 VGPRs of the K=64 sparse-B layout == the dense K=32 operand):
//   lanes 0-15:  col N = lane,    halves h=0..15 -> K = kc*32 + (0..7, 16..23)
//   lanes 16-31: col N = lane-16, halves h=0..15 -> K = kc*32 + (8..15, 24..31)
// index = ((((l*2+mat)*2+kc)*4+nt)*32+lane)*16 + h
// ---------------------------------------------------------------------------
__global__ void pack_w_kernel(const float* __restrict__ Wg,
                              const float* __restrict__ Wb,
                              _Float16* __restrict__ wpack) {
    int idx = blockIdx.x * blockDim.x + threadIdx.x;
    if (idx >= WPACK_ELEMS) return;
    int h    = idx & 15;
    int lane = (idx >> 4) & 31;
    int nt   = (idx >> 9) & 3;
    int kc   = (idx >> 11) & 1;
    int mat  = (idx >> 12) & 1;
    int l    = idx >> 13;
    int k = kc * 32 + ((lane < 16) ? 0 : 8) + ((h < 8) ? h : h + 8);
    int n = nt * 16 + (lane & 15);
    const float* W = (mat == 0) ? Wg : Wb;
    wpack[idx] = (_Float16)W[l * DIM * DIM + k * DIM + n];
}

// ---------------------------------------------------------------------------
// Kernel 3: SpMM via hardware f32 atomics. ego (25.6MB) and side (25.6MB)
// both fit in the 192MB L2, so gathers and atomic scatters are L2-resident.
// One thread per (edge, dim); the edge index is wave-uniform (each wave32
// covers 32 consecutive dims of one edge), so force it into an SGPR with
// readfirstlane -> rows/cols/vals become scalar (s_load) and the vector pipe
// carries only the coalesced ego gather + global_atomic_add_f32 scatter.
// ---------------------------------------------------------------------------
__global__ void spmm_kernel(const int* __restrict__ rows,
                            const int* __restrict__ cols,
                            const float* __restrict__ vals,
                            const float* __restrict__ ego,
                            float* __restrict__ side,
                            int nedges) {
    long long gid = (long long)blockIdx.x * blockDim.x + threadIdx.x;
    int e = __builtin_amdgcn_readfirstlane((int)(gid >> 6));  // wave-uniform
    if (e >= nedges) return;
    int d = (int)(gid & 63);
    int r = rows[e];          // uniform -> scalar load
    int c = cols[e];          // uniform -> scalar load
    float w = vals[e];        // uniform -> scalar load
    float contrib = w * ego[(size_t)c * DIM + d];
#if defined(__HIP_DEVICE_COMPILE__)
    unsafeAtomicAdd(&side[(size_t)r * DIM + d], contrib);  // global_atomic_add_f32
#else
    atomicAdd(&side[(size_t)r * DIM + d], contrib);
#endif
}

// ---------------------------------------------------------------------------
// Kernel 4: fused dense layer. One wave (32 lanes) owns a 16-row node tile:
//   sum_e + bi_e = side@Wg + (ego*side)@Wb + (bg+bb)   via 16x16x32 f16 WMMA
//   ego' = leaky_relu(.., 0.2)  (written back in place, row-local)
//   out  = ego' / max(||row||, eps)  at column offset 64*(layer+1)
// A-matrix f16 layout per lane (ISA A table): row M = lane%16;
//   lanes<16: halves = K(kc*32 + 0..7, +16..23); lanes>=16: K(+8..15, +24..31)
// ---------------------------------------------------------------------------
__global__ __launch_bounds__(128)
void layer_fused_kernel(const float* __restrict__ side,
                        float* __restrict__ ego,
                        const _Float16* __restrict__ wpack,
                        const float* __restrict__ b_gcn,
                        const float* __restrict__ b_bi,
                        float* __restrict__ out,
                        int layer) {
    int wave = threadIdx.x >> 5;
    int lane = threadIdx.x & 31;
    int tile = blockIdx.x * 4 + wave;
    if (tile >= N_NODES / 16) return;

    int row = lane & 15;                 // M (A) / N (B,C,D) index within tile
    int hi  = (lane < 16) ? 0 : 8;
    size_t rbase = ((size_t)tile * 16 + row) * DIM;
    const float* srow = side + rbase;
    const float* erow = ego + rbase;

    // Build A operands: asum = f16(side), abi = f16(ego*side), two K=32 chunks
    v16h asum[2], abi[2];
#pragma unroll
    for (int kc = 0; kc < 2; ++kc) {
        int base = kc * 32 + hi;
        v4f s0 = *(const v4f*)(srow + base);
        v4f s1 = *(const v4f*)(srow + base + 4);
        v4f s2 = *(const v4f*)(srow + base + 16);
        v4f s3 = *(const v4f*)(srow + base + 20);
        v4f e0 = *(const v4f*)(erow + base);
        v4f e1 = *(const v4f*)(erow + base + 4);
        v4f e2 = *(const v4f*)(erow + base + 16);
        v4f e3 = *(const v4f*)(erow + base + 20);
#pragma unroll
        for (int j = 0; j < 4; ++j) {
            asum[kc][j]      = (_Float16)s0[j];
            asum[kc][4 + j]  = (_Float16)s1[j];
            asum[kc][8 + j]  = (_Float16)s2[j];
            asum[kc][12 + j] = (_Float16)s3[j];
            abi[kc][j]       = (_Float16)(s0[j] * e0[j]);
            abi[kc][4 + j]   = (_Float16)(s1[j] * e1[j]);
            abi[kc][8 + j]   = (_Float16)(s2[j] * e2[j]);
            abi[kc][12 + j]  = (_Float16)(s3[j] * e3[j]);
        }
    }

    const v16h* wp = (const v16h*)wpack;
    v8f acc[4];
#pragma unroll
    for (int nt = 0; nt < 4; ++nt) {
        int col = nt * 16 + row;
        float bias = b_gcn[layer * DIM + col] + b_bi[layer * DIM + col];
        v8f c;
#pragma unroll
        for (int v = 0; v < 8; ++v) c[v] = bias;

        v16h bg0 = wp[(((layer * 2 + 0) * 2 + 0) * 4 + nt) * 32 + lane];
        v16h bg1 = wp[(((layer * 2 + 0) * 2 + 1) * 4 + nt) * 32 + lane];
        v16h bb0 = wp[(((layer * 2 + 1) * 2 + 0) * 4 + nt) * 32 + lane];
        v16h bb1 = wp[(((layer * 2 + 1) * 2 + 1) * 4 + nt) * 32 + lane];

        c = __builtin_amdgcn_wmma_f32_16x16x32_f16(false, asum[0], false, bg0,
                                                   (short)0, c, false, false);
        c = __builtin_amdgcn_wmma_f32_16x16x32_f16(false, asum[1], false, bg1,
                                                   (short)0, c, false, false);
        c = __builtin_amdgcn_wmma_f32_16x16x32_f16(false, abi[0], false, bb0,
                                                   (short)0, c, false, false);
        c = __builtin_amdgcn_wmma_f32_16x16x32_f16(false, abi[1], false, bb1,
                                                   (short)0, c, false, false);

#pragma unroll
        for (int v = 0; v < 8; ++v) {
            float x = c[v];
            c[v] = (x > 0.f) ? x : 0.2f * x;   // leaky_relu(0.2)
        }
        acc[nt] = c;
    }

    // Row L2 norms. C/D layout: VGPR v, lanes 0-15 -> row v, lanes 16-31 -> v+8.
    // 16-lane xor reduction (masks 1,2,4,8 stay within each half of the wave).
    float sq[8];
#pragma unroll
    for (int v = 0; v < 8; ++v) {
        sq[v] = acc[0][v] * acc[0][v] + acc[1][v] * acc[1][v] +
                acc[2][v] * acc[2][v] + acc[3][v] * acc[3][v];
    }
#pragma unroll
    for (int m = 1; m < 16; m <<= 1) {
#pragma unroll
        for (int v = 0; v < 8; ++v) sq[v] += __shfl_xor(sq[v], m, 32);
    }
    float inv[8];
#pragma unroll
    for (int v = 0; v < 8; ++v) {
        float nrm = fmaxf(sqrtf(sq[v]), EPS_NRM);
        inv[v] = 1.0f / nrm;
    }

    // Write-back: unnormalized -> ego (next layer input, in place, row-local);
    // normalized -> out at column offset 64*(layer+1).
#pragma unroll
    for (int nt = 0; nt < 4; ++nt) {
        int col = nt * 16 + row;
#pragma unroll
        for (int v = 0; v < 8; ++v) {
            int r = v + hi;
            size_t node = (size_t)tile * 16 + r;
            float x = acc[nt][v];
            ego[node * DIM + col] = x;
            out[node * OUT_DIM + (layer + 1) * DIM + col] = x * inv[v];
        }
    }
}

// ---------------------------------------------------------------------------
extern "C" void kernel_launch(void* const* d_in, const int* in_sizes, int n_in,
                              void* d_out, int out_size, void* d_ws, size_t ws_size,
                              hipStream_t stream) {
    const float* user_emb = (const float*)d_in[0];
    const float* item_emb = (const float*)d_in[1];
    const float* W_gcn    = (const float*)d_in[2];
    const float* b_gcn    = (const float*)d_in[3];
    const float* W_bi     = (const float*)d_in[4];
    const float* b_bi     = (const float*)d_in[5];
    const int*   rows     = (const int*)d_in[6];
    const int*   cols     = (const int*)d_in[7];
    const float* vals     = (const float*)d_in[8];
    const int nedges = in_sizes[6];
    float* out = (float*)d_out;

    const size_t side_bytes  = (size_t)N_NODES * DIM * sizeof(float);   // 25.6 MB
    const size_t wpack_bytes = (size_t)WPACK_ELEMS * sizeof(_Float16);  // 48 KB
    float*    d_side  = (float*)d_ws;
    _Float16* d_wpack = (_Float16*)((char*)d_ws + side_bytes);
    float*    d_ego   = (float*)((char*)d_ws + side_bytes + wpack_bytes);

    {
        int total = N_NODES * DIM;
        init_ego_kernel<<<(total + 255) / 256, 256, 0, stream>>>(user_emb, item_emb,
                                                                 d_ego, out);
    }
    {
        int total = WPACK_ELEMS;
        pack_w_kernel<<<(total + 255) / 256, 256, 0, stream>>>(W_gcn, W_bi, d_wpack);
    }
    for (int l = 0; l < N_LAYERS; ++l) {
        hipMemsetAsync(d_side, 0, side_bytes, stream);
        long long work = (long long)nedges * 64;
        int blocks = (int)((work + 255) / 256);
        spmm_kernel<<<blocks, 256, 0, stream>>>(rows, cols, vals, d_ego, d_side, nedges);
        int ntiles = N_NODES / 16;
        layer_fused_kernel<<<(ntiles + 3) / 4, 128, 0, stream>>>(d_side, d_ego, d_wpack,
                                                                 b_gcn, b_bi, out, l);
    }
    (void)n_in; (void)out_size; (void)ws_size;
}
